// Modelmy_43997644980381
// MI455X (gfx1250) — matrix operand order repaired and checked by reading, not compiled, not run on hardware
//
#include <hip/hip_runtime.h>
#include <hip/hip_bf16.h>
#include <hip/hip_fp16.h>

typedef __attribute__((ext_vector_type(16))) _Float16 v16h;
typedef __attribute__((ext_vector_type(8)))  _Float16 v8h;
typedef __attribute__((ext_vector_type(8)))  float    v8f;

#define HD   300          // hidden size
#define KPAD 640          // 320 (x padded) + 320 (h padded)
#define GPAD 1216         // 4*H=1200 gates padded to 19*64
#define NGG  2700
#define NGP  2720         // NGG padded to multiple of 16 (gates rows)
#define GWW  100
#define NSS  6
#define NWW  654

// ---------------- weight packing ----------------
__global__ void k_pack_w(const float* __restrict__ Wih, const float* __restrict__ Whh,
                         int I, _Float16* __restrict__ wT) {
  int idx = blockIdx.x * blockDim.x + threadIdx.x;
  if (idx >= GPAD * KPAD) return;
  int j = idx / KPAD, k = idx - j * KPAD;
  float v = 0.f;
  if (j < 1200) {
    if (k < 320) { if (k < I) v = Wih[(size_t)j * I + k]; }
    else { int kk = k - 320; if (kk < HD) v = Whh[(size_t)j * HD + kk]; }
  }
  wT[idx] = (_Float16)v;
}

__global__ void k_pack_bias(const float* __restrict__ bih, const float* __restrict__ bhh,
                            float* __restrict__ bias) {
  int j = blockIdx.x * blockDim.x + threadIdx.x;
  if (j >= GPAD) return;
  bias[j] = (j < 1200) ? (bih[j] + bhh[j]) : 0.f;
}

// ---------------- embeddings / lengths ----------------
__global__ void k_embed(const int* __restrict__ toks, const int* __restrict__ pos,
                        const float* __restrict__ glove, const float* __restrict__ pemb,
                        float* __restrict__ xout, int NT) {
  int idx = blockIdx.x * blockDim.x + threadIdx.x;
  if (idx >= NT * 320) return;
  int s = idx / 320, k = idx - s * 320;
  float v = 0.f;
  if (k < 300) v = glove[(size_t)toks[s] * 300 + k];
  else if (k < 308) { int p = pos[s]; if (p != 0) v = pemb[p * 8 + (k - 300)]; }
  xout[idx] = v;
}

__global__ void k_len(const int* __restrict__ toks, int* __restrict__ len,
                      int N, int L, int min1) {
  int n = blockIdx.x * blockDim.x + threadIdx.x;
  if (n >= N) return;
  int c = 0;
  for (int t = 0; t < L; ++t) c += (toks[n * L + t] != 0);
  if (min1 && c < 1) c = 1;
  len[n] = c;
}

__global__ void k_gloss_init(const int* __restrict__ gid, const float* __restrict__ glove,
                             float* __restrict__ xg) {
  int idx = blockIdx.x * blockDim.x + threadIdx.x;
  if (idx >= NGG * GWW * 320) return;
  int s = idx / 320, k = idx - s * 320;
  float v = 0.f;
  if (k < 300) v = glove[(size_t)gid[s] * 300 + k];
  xg[idx] = v;
}

__global__ void k_zero(float* __restrict__ p, int n) {
  int i = blockIdx.x * blockDim.x + threadIdx.x;
  if (i < n) p[i] = 0.f;
}

// ---------------- WMMA LSTM step GEMM: gates = [x_t | h] @ WcombT + bias ----------------
// gates buffer has NPAD (=ceil16(N)) rows; stores are unconditional (rows >= N hold garbage
// that the cell kernel never reads; LDS A rows >= N are zero so WMMA math stays clean).
__global__ void __launch_bounds__(128)
k_gemm_step(const float* __restrict__ X, const float* __restrict__ Hst,
            const _Float16* __restrict__ wT, const float* __restrict__ bias,
            float* __restrict__ gates, int N, int T, int t) {
  __shared__ __align__(16) _Float16 As[16 * 648];   // 16 rows x 640 K, row pitch 648 (16B aligned)
  const int tid   = threadIdx.x;
  const int mbase = blockIdx.y * 16;

  // stage A strip ([x_t | h]) into LDS as f16; rows beyond N are zero
  for (int idx = tid; idx < 16 * KPAD; idx += 128) {
    int r = idx / KPAD, k = idx - r * KPAD;
    int n = mbase + r;
    float v = 0.f;
    if (n < N) {
      if (k < 320) v = X[((size_t)n * T + t) * 320 + k];
      else { int kk = k - 320; if (kk < HD) v = Hst[(size_t)n * HD + kk]; }
    }
    As[r * 648 + k] = (_Float16)v;
  }
  __builtin_amdgcn_s_wait_tensorcnt(0);   // CDNA5 split-counter wait (no-op here)
  __syncthreads();

  const int wave = tid >> 5, lane = tid & 31;
  const int col  = lane & 15, half = lane >> 4;
  const int gbase = blockIdx.x * 64 + wave * 16;
  const _Float16* __restrict__ wrow = wT + (size_t)(gbase + col) * KPAD;
  const _Float16* __restrict__ arow = As + col * 648;   // A row m = lane%16

  // fold bias into the accumulator: D lane column == gate index, same for all 8 M-rows
  const float bv = bias[gbase + col];
  v8f acc = {bv, bv, bv, bv, bv, bv, bv, bv};

  #pragma unroll 4
  for (int k0 = 0; k0 < KPAD; k0 += 32) {
    // A 16x32 f16 layout: lanes0-15 K{0..7,16..23}, lanes16-31 K{8..15,24..31}
    v8h alo = *(const v8h*)(arow + k0 + half * 8);
    v8h ahi = *(const v8h*)(arow + k0 + 16 + half * 8);
    // B 32x16 f16 layout: lanes0-15 K{0..7,16..23}, lanes16-31 K{8..15,24..31} (matches A)
    const _Float16* wp = wrow + k0 + half * 8;
    v8h blo = *(const v8h*)(wp);
    v8h bhi = *(const v8h*)(wp + 16);
    __builtin_prefetch(wp + 32, 0, 1);    // global_prefetch_b8 on next K-chunk of weights
    v16h a = __builtin_shufflevector(alo, ahi, 0,1,2,3,4,5,6,7,8,9,10,11,12,13,14,15);
    v16h b = __builtin_shufflevector(blo, bhi, 0,1,2,3,4,5,6,7,8,9,10,11,12,13,14,15);
    acc = __builtin_amdgcn_wmma_f32_16x16x32_f16(false, a, false, b, (short)0, acc, false, false);
  }

  // D layout: lane n = lane%16, VGPR r -> M = half*8 + r; unconditional (gates rows padded)
  float* __restrict__ gout = gates + (size_t)(mbase + half * 8) * GPAD + gbase + col;
  #pragma unroll
  for (int r = 0; r < 8; ++r) gout[(size_t)r * GPAD] = acc[r];
}

// ---------------- LSTM cell pointwise (bias already in gates) ----------------
__global__ void k_cell(const float* __restrict__ gates,
                       float* __restrict__ h, float* __restrict__ c,
                       const int* __restrict__ len, int N, int t) {
  int idx = blockIdx.x * blockDim.x + threadIdx.x;
  if (idx >= N * HD) return;
  int n = idx / HD, j = idx - n * HD;
  const float* gr = gates + (size_t)n * GPAD;
  float gi = gr[j];
  float gf = gr[j + 300];
  float gg = gr[j + 600];
  float go = gr[j + 900];
  float si = 1.f / (1.f + __expf(-gi));
  float sf = 1.f / (1.f + __expf(-gf));
  float so = 1.f / (1.f + __expf(-go));
  float cn = sf * c[idx] + si * tanhf(gg);
  float hn = so * tanhf(cn);
  if (t < len[n]) { c[idx] = cn; h[idx] = hn; }
}

// ---------------- alpha normalization (alpha2 == alpha1) ----------------
__global__ void k_alpha1(const float* __restrict__ alpha, float* __restrict__ a1) {
  int row = blockIdx.x * blockDim.x + threadIdx.x;
  if (row >= NGG * NSS) return;
  const float* ar = alpha + (size_t)row * NSS;
  float s = 0.f;
  for (int i = 0; i < NSS; ++i) s += ar[i];
  float inv = (s == 0.f) ? 0.f : 1.f / s;
  for (int i = 0; i < NSS; ++i) a1[(size_t)row * NSS + i] = ar[i] * inv;
}

// ---------------- graph propagation ----------------
__global__ void k_bgather(const float* __restrict__ g, const int* __restrict__ s2g,
                          const int* __restrict__ wts, float* __restrict__ bbuf) {
  int idx = blockIdx.x * blockDim.x + threadIdx.x;
  if (idx >= NWW * NSS * HD) return;
  int k = idx % HD, ws_ = idx / HD;
  float v = 0.f;
  int a = wts[ws_];
  if (a >= 1) { int sg = s2g[a - 1]; if (sg >= 1) v = g[(size_t)(sg - 1) * HD + k]; }
  bbuf[idx] = v;
}

__global__ void k_dcompute(const float* __restrict__ bbuf, const int* __restrict__ g2w,
                           const float* __restrict__ a1, float* __restrict__ dbuf) {
  int idx = blockIdx.x * blockDim.x + threadIdx.x;
  if (idx >= NGG * NSS * HD) return;
  int k = idx % HD, nj = idx / HD;
  int n = nj / NSS, j = nj - n * NSS;
  int gw = g2w[nj];
  float v = 0.f;
  if (gw >= 1) {
    const float* br = bbuf + (size_t)(gw - 1) * NSS * HD + k;
    const float* al = a1 + (size_t)n * NSS * NSS + j * NSS;
    #pragma unroll
    for (int s = 0; s < NSS; ++s) v += al[s] * br[s * HD];
  }
  dbuf[idx] = v;
}

__global__ void k_update_inputg(const float* __restrict__ dbuf, const int* __restrict__ gmask,
                                float* __restrict__ xg) {
  int idx = blockIdx.x * blockDim.x + threadIdx.x;
  if (idx >= NGG * GWW * HD) return;
  int k = idx % HD, nt = idx / HD;
  int n = nt / GWW;
  int m = gmask[nt];
  if (m != 0) xg[(size_t)nt * 320 + k] = dbuf[((size_t)n * NSS + (m - 1)) * HD + k];
}

// ---------------- gloss match + gather ----------------
__global__ void __launch_bounds__(256)
k_match(const int* __restrict__ gl2, const int* __restrict__ gid,
        const float* __restrict__ og, float* __restrict__ out) {
  __shared__ int best;
  int r = blockIdx.x;
  if (threadIdx.x == 0) best = 0;
  __syncthreads();
  const int* q = gl2 + (size_t)r * GWW;
  for (int n = threadIdx.x; n < NGG; n += 256) {
    const int* p = gid + (size_t)n * GWW;
    bool ok = true;
    for (int t = 0; t < GWW; ++t) if (q[t] != p[t]) { ok = false; break; }
    if (ok) atomicMax(&best, n + 1);
  }
  __syncthreads();
  int b = best;
  for (int k = threadIdx.x; k < HD; k += 256)
    out[(size_t)r * HD + k] = b ? og[(size_t)(b - 1) * HD + k] : 0.f;
}

// ---------------- small output helpers ----------------
__global__ void k_copy(float* __restrict__ dst, const float* __restrict__ src, int n) {
  int i = blockIdx.x * blockDim.x + threadIdx.x;
  if (i < n) dst[i] = src[i];
}
__global__ void k_max2(float* __restrict__ dst, const float* __restrict__ a,
                       const float* __restrict__ b, int n) {
  int i = blockIdx.x * blockDim.x + threadIdx.x;
  if (i < n) dst[i] = fmaxf(a[i], b[i]);
}
__global__ void k_i2f(float* __restrict__ dst, const int* __restrict__ src, int n) {
  int i = blockIdx.x * blockDim.x + threadIdx.x;
  if (i < n) dst[i] = (float)src[i];
}

// =============================================================================
extern "C" void kernel_launch(void* const* d_in, const int* in_sizes, int n_in,
                              void* d_out, int out_size, void* d_ws, size_t ws_size,
                              hipStream_t stream) {
  const int*   inputs_f = (const int*)d_in[0];
  const int*   inputs_b = (const int*)d_in[1];
  const int*   sense_ids = (const int*)d_in[2];
  const int*   glosses  = (const int*)d_in[3];
  const float* sense_masks = (const float*)d_in[4];
  const int*   pos_f = (const int*)d_in[5];
  const int*   pos_b = (const int*)d_in[6];
  const float* glove = (const float*)d_in[7];
  const float* pemb  = (const float*)d_in[8];
  const float* alpha = (const float*)d_in[9];
  const int*   gloss_id = (const int*)d_in[10];
  const int*   s2g = (const int*)d_in[11];
  const int*   wts = (const int*)d_in[12];
  const int*   g2w = (const int*)d_in[13];
  const int*   gmask = (const int*)d_in[14];
  const float* Wih0 = (const float*)d_in[15]; const float* Whh0 = (const float*)d_in[16];
  const float* bih0 = (const float*)d_in[17]; const float* bhh0 = (const float*)d_in[18];
  const float* Wih1 = (const float*)d_in[19]; const float* Whh1 = (const float*)d_in[20];
  const float* bih1 = (const float*)d_in[21]; const float* bhh1 = (const float*)d_in[22];
  const float* Wih2 = (const float*)d_in[23]; const float* Whh2 = (const float*)d_in[24];
  const float* bih2 = (const float*)d_in[25]; const float* bhh2 = (const float*)d_in[26];
  float* out = (float*)d_out;

  // ---- workspace carve ----
  char* ws = (char*)d_ws;
  size_t off = 0;
  auto take = [&](size_t bytes) {
    size_t o = off; off = (off + bytes + 255) & ~(size_t)255; return o;
  };
  _Float16* wT0 = (_Float16*)(ws + take((size_t)GPAD * KPAD * 2));
  _Float16* wT1 = (_Float16*)(ws + take((size_t)GPAD * KPAD * 2));
  _Float16* wT2 = (_Float16*)(ws + take((size_t)GPAD * KPAD * 2));
  float* bia0 = (float*)(ws + take(GPAD * 4));
  float* bia1 = (float*)(ws + take(GPAD * 4));
  float* bia2 = (float*)(ws + take(GPAD * 4));
  float* xf = (float*)(ws + take((size_t)32 * 64 * 320 * 4));
  float* xb = (float*)(ws + take((size_t)32 * 64 * 320 * 4));
  int* lenf = (int*)(ws + take(32 * 4));
  int* lenb = (int*)(ws + take(32 * 4));
  float* xg = (float*)(ws + take((size_t)NGG * GWW * 320 * 4));
  int* glen = (int*)(ws + take(NGG * 4));
  float* gates = (float*)(ws + take((size_t)NGP * GPAD * 4));   // padded rows
  float* hbuf = (float*)(ws + take((size_t)NGG * HD * 4));
  float* cbuf = (float*)(ws + take((size_t)NGG * HD * 4));
  float* fwbuf = (float*)(ws + take((size_t)32 * HD * 4));
  float* a1 = (float*)(ws + take((size_t)NGG * NSS * NSS * 4));
  float* bbuf = (float*)(ws + take((size_t)NWW * NSS * HD * 4));
  float* dbuf = (float*)(ws + take((size_t)NGG * NSS * HD * 4));

  auto NB = [](long n) { return dim3((unsigned)((n + 255) / 256)); };

  // ---- prep ----
  k_pack_w<<<NB((long)GPAD * KPAD), 256, 0, stream>>>(Wih0, Whh0, 308, wT0);
  k_pack_w<<<NB((long)GPAD * KPAD), 256, 0, stream>>>(Wih1, Whh1, 308, wT1);
  k_pack_w<<<NB((long)GPAD * KPAD), 256, 0, stream>>>(Wih2, Whh2, 300, wT2);
  k_pack_bias<<<NB(GPAD), 256, 0, stream>>>(bih0, bhh0, bia0);
  k_pack_bias<<<NB(GPAD), 256, 0, stream>>>(bih1, bhh1, bia1);
  k_pack_bias<<<NB(GPAD), 256, 0, stream>>>(bih2, bhh2, bia2);
  k_embed<<<NB((long)2048 * 320), 256, 0, stream>>>(inputs_f, pos_f, glove, pemb, xf, 2048);
  k_embed<<<NB((long)2048 * 320), 256, 0, stream>>>(inputs_b, pos_b, glove, pemb, xb, 2048);
  k_len<<<NB(32), 256, 0, stream>>>(inputs_f, lenf, 32, 64, 1);
  k_len<<<NB(32), 256, 0, stream>>>(inputs_b, lenb, 32, 64, 1);
  k_gloss_init<<<NB((long)NGG * GWW * 320), 256, 0, stream>>>(gloss_id, glove, xg);
  k_len<<<NB(NGG), 256, 0, stream>>>(gloss_id, glen, NGG, GWW, 0);
  k_alpha1<<<NB((long)NGG * NSS), 256, 0, stream>>>(alpha, a1);

  auto run_lstm = [&](const float* X, int N, int T, const _Float16* wT,
                      const float* bias, const int* len) {
    long hc = (long)N * HD;
    k_zero<<<NB(hc), 256, 0, stream>>>(hbuf, (int)hc);
    k_zero<<<NB(hc), 256, 0, stream>>>(cbuf, (int)hc);
    dim3 gg(GPAD / 64, (N + 15) / 16);
    for (int t = 0; t < T; ++t) {
      k_gemm_step<<<gg, 128, 0, stream>>>(X, hbuf, wT, bias, gates, N, T, t);
      k_cell<<<NB(hc), 256, 0, stream>>>(gates, hbuf, cbuf, len, N, t);
    }
  };

  // ---- sentence LSTMs: fw, bw, sentence = max(fw, bw) ----
  run_lstm(xf, 32, 64, wT0, bia0, lenf);
  k_copy<<<NB(9600), 256, 0, stream>>>(fwbuf, hbuf, 9600);
  run_lstm(xb, 32, 64, wT1, bia1, lenb);
  k_max2<<<NB(9600), 256, 0, stream>>>(out + 0, fwbuf, hbuf, 9600);

  // ---- passthrough outputs ----
  k_i2f<<<NB(32), 256, 0, stream>>>(out + 9600, sense_ids, 32);
  k_copy<<<NB(192), 256, 0, stream>>>(out + 67232, sense_masks, 192);
  k_copy<<<NB((long)NGG * 36), 256, 0, stream>>>(out + 877424, a1, NGG * 36);  // alpha2 == alpha1

  // ---- gloss LSTM: 3 propagation iterations + final ----
  for (int p = 0; p < 4; ++p) {
    run_lstm(xg, NGG, GWW, wT2, bia2, glen);
    if (p < 3) {
      k_bgather<<<NB((long)NWW * NSS * HD), 256, 0, stream>>>(hbuf, s2g, wts, bbuf);
      k_dcompute<<<NB((long)NGG * NSS * HD), 256, 0, stream>>>(bbuf, g2w, a1, dbuf);
      k_update_inputg<<<NB((long)NGG * GWW * HD), 256, 0, stream>>>(dbuf, gmask, xg);
    }
  }

  // ---- output_g and all_gloss ----
  k_copy<<<NB((long)NGG * HD), 256, 0, stream>>>(out + 67424, hbuf, NGG * HD);
  k_match<<<dim3(192), 256, 0, stream>>>(glosses, gloss_id, hbuf, out + 9632);
}
